// TanhBasisActivationLayer_56495999811651
// MI455X (gfx1250) — matrix operand order repaired and checked by reading, not compiled, not run on hardware
//
#include <hip/hip_runtime.h>
#include <stdint.h>

// ---------------------------------------------------------------------------
// TanhBasisActivationLayer for MI455X (gfx1250, wave32, WMMA + async-to-LDS)
//
//   y[b,j] = sum_{i,m} coeffs[j,i,m] * sigmoid(2*gamma*(x[b,i]-c[i,m]))
//
//   A (8192 x 4096, f16) = basis matrix (workspace)
//   B (512  x 4096, f16) = coeffs cast to f16 (layout already K-contiguous)
//   C = A @ B^T via v_wmma_f32_16x16x32_f16, tiles staged through LDS with
//   global_load_async_to_lds_b128 (ASYNCcnt), double-buffered.
// ---------------------------------------------------------------------------

typedef __attribute__((ext_vector_type(16))) _Float16 v16h;
typedef __attribute__((ext_vector_type(8)))  _Float16 v8h;
typedef __attribute__((ext_vector_type(4)))  _Float16 v4h;
typedef __attribute__((ext_vector_type(8)))  float    v8f;
typedef __attribute__((ext_vector_type(4)))  float    v4f;

#define IN_COUNT  512
#define OUT_COUNT 512
#define NUM_BASIS 8
#define BATCH     8192
#define K_DIM     (IN_COUNT * NUM_BASIS)   // 4096

#define KSTAGE    64                        // K per LDS stage
#define A_LD      72                        // padded LDS row pitch (halfs)
#define TILE_MN   128                       // rows in each LDS tile
#define BUF_HALFS (2 * TILE_MN * A_LD)      // A-tile + B-tile per buffer

// ---------------------------------------------------------------------------
// Kernel 0: coeffs f32 -> f16 (pure cast; memory layout is already B^T[n][k])
// ---------------------------------------------------------------------------
__global__ __launch_bounds__(256) void coeffs_to_f16_kernel(
    const float* __restrict__ coeffs, _Float16* __restrict__ B)
{
    int idx = (blockIdx.x * 256 + threadIdx.x) * 4;
    v4f v = *(const v4f*)(coeffs + idx);
    v4h o;
    o[0] = (_Float16)v[0];
    o[1] = (_Float16)v[1];
    o[2] = (_Float16)v[2];
    o[3] = (_Float16)v[3];
    *(v4h*)(B + idx) = o;
}

// ---------------------------------------------------------------------------
// Kernel 1: basis matrix A[b*512 + i][m] = sigmoid(2*slope*(x - center)), f16
// ---------------------------------------------------------------------------
__global__ __launch_bounds__(256) void basis_f16_kernel(
    const float* __restrict__ x,
    const float* __restrict__ centers,   // (IN_COUNT, NUM_BASIS)
    const float* __restrict__ slopes,    // (IN_COUNT, NUM_BASIS)
    v8h* __restrict__ A)                 // (BATCH*IN_COUNT) rows of 8 halfs
{
    int idx = blockIdx.x * 256 + threadIdx.x;      // flat (b,i)
    int i   = idx & (IN_COUNT - 1);
    float xv = x[idx];

    const float* cb = centers + i * NUM_BASIS;
    const float* sb = slopes  + i * NUM_BASIS;

    v8h out;
#pragma unroll
    for (int m = 0; m < NUM_BASIS; ++m) {
        float z = sb[m] * (xv - cb[m]);
        // 0.5*(1+tanh(z)) == 1/(1+exp(-2z)) ; exp via hardware v_exp_f32
        float e = __builtin_amdgcn_exp2f(z * (-2.0f * 1.4426950408889634f));
        float s = 1.0f / (1.0f + e);
        out[m] = (_Float16)s;
    }
    A[idx] = out;
}

// ---------------------------------------------------------------------------
// Kernel 2: C = A @ B^T. 256 threads = 8 waves, 128x128 block tile.
// Wave (wm 0..3, wn 0..1) owns a 32x64 sub-tile => 2x4 fragments.
// Tiles staged via global_load_async_to_lds_b128, double-buffered.
// ---------------------------------------------------------------------------
__global__ __launch_bounds__(256) void wmma_gemm_f16_kernel(
    const _Float16* __restrict__ A,   // [BATCH][K_DIM] row-major
    const _Float16* __restrict__ B,   // [OUT_COUNT][K_DIM] row-major (B^T)
    float* __restrict__ C)            // [BATCH][OUT_COUNT]
{
    __shared__ _Float16 smem[2 * BUF_HALFS];   // double-buffered A+B tiles

    const int tid   = threadIdx.x;
    const int lane  = tid & 31;
    const int wave  = tid >> 5;
    const int wm    = wave & 3;        // M strip within block: 4 x 32 rows
    const int wn    = wave >> 2;       // N strip within block: 2 x 64 cols
    const int l15   = lane & 15;
    const int lhalf = lane >> 4;       // 0: lanes 0-15, 1: lanes 16-31

    const int blockM = blockIdx.x * 128;
    const int blockN = blockIdx.y * 128;

    const _Float16* gA = A + (size_t)blockM * K_DIM;
    const _Float16* gB = B + (size_t)blockN * K_DIM;

    // LDS base as a 32-bit LDS byte address (low 32 bits of generic addr).
    const unsigned ldsBase = (unsigned)(uintptr_t)&smem[0];

    // Per-thread async staging assignment: each thread moves 4 b128 chunks
    // of the A tile and 4 of the B tile (128 rows x 8 chunks per tile).
    const int srow = tid >> 1;            // staging row 0..127
    const int sch  = (tid & 1) * 4;       // first chunk (1 chunk = 8 halfs)

    const _Float16* gArow = gA + (size_t)srow * K_DIM + sch * 8;
    const _Float16* gBrow = gB + (size_t)srow * K_DIM + sch * 8;
    const unsigned  lArow = (unsigned)((srow * A_LD + sch * 8) * 2);
    const unsigned  lBrow = lArow + (unsigned)(TILE_MN * A_LD * 2);

    #define ISSUE_STAGE(bufsel, kofs)                                          \
        do {                                                                   \
            unsigned _la = ldsBase + (unsigned)((bufsel) * BUF_HALFS * 2) + lArow; \
            unsigned _lb = ldsBase + (unsigned)((bufsel) * BUF_HALFS * 2) + lBrow; \
            const _Float16* _ga = gArow + (kofs);                              \
            const _Float16* _gb = gBrow + (kofs);                              \
            _Pragma("unroll")                                                  \
            for (int _c = 0; _c < 4; ++_c) {                                   \
                asm volatile("global_load_async_to_lds_b128 %0, %1, off"       \
                    :: "v"(_la + _c * 16),                                     \
                       "v"((unsigned long long)(uintptr_t)(_ga + _c * 8))      \
                    : "memory");                                               \
                asm volatile("global_load_async_to_lds_b128 %0, %1, off"       \
                    :: "v"(_lb + _c * 16),                                     \
                       "v"((unsigned long long)(uintptr_t)(_gb + _c * 8))      \
                    : "memory");                                               \
            }                                                                  \
        } while (0)

    v8f acc[2][4];
#pragma unroll
    for (int mi = 0; mi < 2; ++mi)
#pragma unroll
        for (int ni = 0; ni < 4; ++ni)
            acc[mi][ni] = (v8f)0.0f;

    // Fragment base indices inside an LDS stage (16-bit WMMA layouts):
    //  A: lanes 0-15 hold K {0..7,16..23}, lanes 16-31 K {8..15,24..31}
    //  B: lanes 0-15 hold K {0..7,16..23} of col N=lane, lanes 16-31 K {8..15,24..31}
    const int aBase0 = (wm * 32 +  0 + l15) * A_LD + lhalf * 8;
    const int aBase1 = (wm * 32 + 16 + l15) * A_LD + lhalf * 8;
    int bBase[4];
#pragma unroll
    for (int ni = 0; ni < 4; ++ni)
        bBase[ni] = TILE_MN * A_LD + (wn * 64 + ni * 16 + l15) * A_LD + lhalf * 8;

    // Prologue: stage 0 into buffer 0.
    ISSUE_STAGE(0, 0);

    const int NSTAGE = K_DIM / KSTAGE;     // 64
    for (int ks = 0; ks < NSTAGE; ++ks) {
        const int buf  = ks & 1;
        const int boff = buf * BUF_HALFS;

        asm volatile("s_wait_asynccnt 0" ::: "memory");
        __syncthreads();                    // stage ks visible to all waves

        if (ks + 1 < NSTAGE)                // overlap DMA of next stage
            ISSUE_STAGE(buf ^ 1, (ks + 1) * KSTAGE);

#pragma unroll
        for (int ksub = 0; ksub < KSTAGE; ksub += 32) {
            v16h afrag[2];
            {
                v8h lo0 = *(const v8h*)&smem[boff + aBase0 + ksub];
                v8h hi0 = *(const v8h*)&smem[boff + aBase0 + ksub + 16];
                afrag[0] = __builtin_shufflevector(lo0, hi0,
                            0,1,2,3,4,5,6,7, 8,9,10,11,12,13,14,15);
                v8h lo1 = *(const v8h*)&smem[boff + aBase1 + ksub];
                v8h hi1 = *(const v8h*)&smem[boff + aBase1 + ksub + 16];
                afrag[1] = __builtin_shufflevector(lo1, hi1,
                            0,1,2,3,4,5,6,7, 8,9,10,11,12,13,14,15);
            }
            v16h bfrag[4];
#pragma unroll
            for (int ni = 0; ni < 4; ++ni) {
                v8h lo = *(const v8h*)&smem[boff + bBase[ni] + ksub];
                v8h hi = *(const v8h*)&smem[boff + bBase[ni] + ksub + 16];
                bfrag[ni] = __builtin_shufflevector(lo, hi,
                            0,1,2,3,4,5,6,7, 8,9,10,11,12,13,14,15);
            }
#pragma unroll
            for (int mi = 0; mi < 2; ++mi)
#pragma unroll
                for (int ni = 0; ni < 4; ++ni)
                    acc[mi][ni] = __builtin_amdgcn_wmma_f32_16x16x32_f16(
                        false, afrag[mi], false, bfrag[ni],
                        (short)0, acc[mi][ni], false, false);
        }

        __syncthreads();   // all waves done reading buf before it is refilled
    }
    #undef ISSUE_STAGE

    // Epilogue: f32 C/D layout — VGPR e holds row e (lanes 0-15) or row e+8
    // (lanes 16-31); column = blockN + strip + (lane&15).
    const int rowbase = blockM + wm * 32 + lhalf * 8;
    const int col     = blockN + wn * 64 + l15;
#pragma unroll
    for (int mi = 0; mi < 2; ++mi) {
#pragma unroll
        for (int ni = 0; ni < 4; ++ni) {
#pragma unroll
            for (int e = 0; e < 8; ++e) {
                C[(size_t)(rowbase + mi * 16 + e) * OUT_COUNT
                  + col + ni * 16] = acc[mi][ni][e];
            }
        }
    }
}

// ---------------------------------------------------------------------------
// Launch: convert coeffs, materialize basis, run WMMA GEMM.
// Workspace: A = 8192*4096*2 = 64 MiB, B = 512*4096*2 = 4 MiB.
// ---------------------------------------------------------------------------
extern "C" void kernel_launch(void* const* d_in, const int* in_sizes, int n_in,
                              void* d_out, int out_size, void* d_ws, size_t ws_size,
                              hipStream_t stream) {
    const float* x       = (const float*)d_in[0];
    const float* coeffs  = (const float*)d_in[1];
    const float* centers = (const float*)d_in[2];
    const float* slopes  = (const float*)d_in[3];
    float*       out     = (float*)d_out;

    _Float16* A = (_Float16*)d_ws;
    _Float16* B = A + (size_t)BATCH * K_DIM;

    // coeffs: 512*512*8 = 2,097,152 f32 -> f16 (4 per thread)
    coeffs_to_f16_kernel<<<(OUT_COUNT * K_DIM / 4) / 256, 256, 0, stream>>>(coeffs, B);

    // basis: one thread per (b,i) = 4,194,304 threads
    basis_f16_kernel<<<(BATCH * IN_COUNT) / 256, 256, 0, stream>>>(
        x, centers, slopes, (v8h*)A);

    // GEMM: 64 x 4 blocks of 128x128 output tile
    dim3 grid(BATCH / 128, OUT_COUNT / 128);
    wmma_gemm_f16_kernel<<<grid, 256, 0, stream>>>(A, B, out);
}
